// DifferentiableHardKMeans_63917703299104
// MI455X (gfx1250) — matrix operand order repaired and checked by reading, not compiled, not run on hardware
//
#include <hip/hip_runtime.h>
#include <hip/hip_bf16.h>

// ---------------------------------------------------------------------------
// DifferentiableHardKMeans for MI455X (gfx1250, wave32, WMMA)
//
// B=8, N=65536, D=256, K=64.  Memory-bound target: ~780MB @ 23.3TB/s ~= 33us.
// GEMM xc = X * C^T done in split-bf16 (hi+lo) via v_wmma_f32_16x16x32_bf16,
// 3 WMMAs per tile (hi*hi + hi*lo + lo*hi) for ~fp32 accuracy, fp32 accum.
//
// Round-1 fix: the fully-unrolled k-loop caused VGPR spills to scratch inside
// the WMMA loop (scratch_load_b128 reloads in the asm).  The k-loop is now
// kept rolled (#pragma unroll 1) so B fragments / A conversions stay transient
// (~100 VGPRs peak) and no spill traffic competes with the 23.3 TB/s stream.
// ---------------------------------------------------------------------------

typedef __attribute__((ext_vector_type(16))) __bf16 v16bf;
typedef __attribute__((ext_vector_type(8)))  __bf16 v8bf;
typedef __attribute__((ext_vector_type(8)))  float  v8f;
typedef __attribute__((ext_vector_type(4)))  float  v4f;

#define TAU_F 0.1f

constexpr int  Dd             = 256;
constexpr int  Kc             = 64;
constexpr long ROWS           = 8L * 65536L;        // B*N = 524288 pixel rows
constexpr int  ROWS_PER_BLOCK = 128;                // 8 waves * 16 rows
constexpr int  NTILES         = (int)(ROWS / ROWS_PER_BLOCK);  // 4096
constexpr int  CPAD           = 264;                // bf16 row stride (528B) -> LDS conflict-free

__global__ __launch_bounds__(256)
void kmeans_hard_assign(const float* __restrict__ X,   // [ROWS, 256]
                        const float* __restrict__ C,   // [64, 256]
                        const float* __restrict__ G,   // [ROWS, 64] gumbel noise
                        float* __restrict__ Out)       // [ROWS, 64]
{
    // Centers in split-bf16 form + ||c||^2, built once per block.
    __shared__ __bf16 sCH[Kc * CPAD];
    __shared__ __bf16 sCL[Kc * CPAD];
    __shared__ float  sC2[Kc];

    const int tid = threadIdx.x;
    if (tid < Kc) sC2[tid] = 0.0f;
    __syncthreads();

    for (int i = tid; i < Kc * Dd; i += 256) {
        const int n  = i >> 8;      // cluster
        const int kd = i & 255;     // dim
        const float f = C[i];
        const __bf16 h = (__bf16)f;
        const __bf16 l = (__bf16)(f - (float)h);
        sCH[n * CPAD + kd] = h;
        sCL[n * CPAD + kd] = l;
        atomicAdd(&sC2[n], f * f);  // ds_add_f32
    }
    __syncthreads();

    const int wave = tid >> 5;
    const int lane = tid & 31;
    const int m    = lane & 15;     // A row / output column-within-tile
    const int half = lane >> 4;     // lane group 0/1

    for (int tile = blockIdx.x; tile < NTILES; tile += gridDim.x) {
        const long row0 = (long)tile * ROWS_PER_BLOCK + (long)wave * 16;
        const float* xrow = X + (row0 + m) * Dd;

        // Prefetch next tile's pixel rows while we crunch this one.
        const int ntile = tile + gridDim.x;
        if (ntile < NTILES) {
            const long nrow0 = (long)ntile * ROWS_PER_BLOCK + (long)wave * 16;
            __builtin_prefetch(X + (nrow0 + m) * Dd, 0, 0);
        }

        v8f acc[4] = {};            // 16x64 fp32 accumulators (4 N-tiles of 16)
        float x2p = 0.0f;           // partial ||x||^2 for row m (this lane's K subset)

        // Rolled on purpose: full unroll spilled B fragments to scratch.
        #pragma unroll 1
        for (int kk = 0; kk < 8; ++kk) {            // D = 8 * 32
            // A fragment: 16-bit A 16x32 layout -> half 0: K {0-7,16-23}, half 1: {8-15,24-31}
            const int kb = kk * 32 + half * 8;
            const v4f a0 = *(const v4f*)(xrow + kb);
            const v4f a1 = *(const v4f*)(xrow + kb + 4);
            const v4f a2 = *(const v4f*)(xrow + kb + 16);
            const v4f a3 = *(const v4f*)(xrow + kb + 20);

            v16bf ah, al;
            #pragma unroll
            for (int j = 0; j < 4; ++j) {
                const float f0 = a0[j], f1 = a1[j], f2 = a2[j], f3 = a3[j];
                x2p += f0 * f0 + f1 * f1 + f2 * f2 + f3 * f3;
                __bf16 h;
                h = (__bf16)f0; ah[j]      = h; al[j]      = (__bf16)(f0 - (float)h);
                h = (__bf16)f1; ah[4 + j]  = h; al[4 + j]  = (__bf16)(f1 - (float)h);
                h = (__bf16)f2; ah[8 + j]  = h; al[8 + j]  = (__bf16)(f2 - (float)h);
                h = (__bf16)f3; ah[12 + j] = h; al[12 + j] = (__bf16)(f3 - (float)h);
            }

            // B fragment: 32x16 K-major -> half 0 holds K {0-7,16-23}, half 1 holds {8-15,24-31}
            const int cb = kk * 32 + half * 8;
            #pragma unroll
            for (int t = 0; t < 4; ++t) {
                const int crow = (t * 16 + m) * CPAD + cb;
                const v8bf bh0 = *(const v8bf*)(sCH + crow);
                const v8bf bh1 = *(const v8bf*)(sCH + crow + 16);
                const v8bf bl0 = *(const v8bf*)(sCL + crow);
                const v8bf bl1 = *(const v8bf*)(sCL + crow + 16);
                v16bf bh, bl;
                #pragma unroll
                for (int j = 0; j < 8; ++j) {
                    bh[j] = bh0[j]; bh[8 + j] = bh1[j];
                    bl[j] = bl0[j]; bl[8 + j] = bl1[j];
                }
                // xc ~= xh*ch + xh*cl + xl*ch  (fp32 accumulate)
                acc[t] = __builtin_amdgcn_wmma_f32_16x16x32_bf16(
                    false, ah, false, bh, (short)0, acc[t], false, false);
                acc[t] = __builtin_amdgcn_wmma_f32_16x16x32_bf16(
                    false, ah, false, bl, (short)0, acc[t], false, false);
                acc[t] = __builtin_amdgcn_wmma_f32_16x16x32_bf16(
                    false, al, false, bh, (short)0, acc[t], false, false);
            }
        }

        // Lane L and L+16 each summed half of row m's squares -> combine.
        x2p += __shfl_xor(x2p, 16, 32);   // now: full ||x||^2 of row (lane&15)

        // ------------------ epilogue: distance, gumbel softmax, hard assign ----
        #pragma unroll
        for (int r = 0; r < 8; ++r) {
            const int  rr   = r + half * 8;         // output row within 16-tile
            const long orow = row0 + rr;
            const float xx2 = __shfl(x2p, rr, 32);  // ||x||^2 of that row

            float z[4], e[4];
            #pragma unroll
            for (int t = 0; t < 4; ++t) {
                const int n = t * 16 + m;
                const float xc = acc[t][r];
                float d2 = xx2 + sC2[n] - 2.0f * xc;
                d2 = fmaxf(d2, 0.0f);
                const float dist = sqrtf(d2);
                const float g = G[orow * Kc + n];
                z[t] = (g - dist) / TAU_F;
            }

            // argmax over 64 (first-max-wins tie break, like jnp.argmax)
            float bv = z[0];
            int   bi = m;
            #pragma unroll
            for (int t = 1; t < 4; ++t) {
                const int n = t * 16 + m;
                if (z[t] > bv) { bv = z[t]; bi = n; }
            }
            #pragma unroll
            for (int off = 1; off < 16; off <<= 1) {
                const float ov = __shfl_xor(bv, off, 32);
                const int   oi = __shfl_xor(bi, off, 32);
                if (ov > bv || (ov == bv && oi < bi)) { bv = ov; bi = oi; }
            }

            // stable softmax: exp(z - max) / sum
            float s = 0.0f;
            #pragma unroll
            for (int t = 0; t < 4; ++t) { e[t] = expf(z[t] - bv); s += e[t]; }
            #pragma unroll
            for (int off = 1; off < 16; off <<= 1) s += __shfl_xor(s, off, 32);

            #pragma unroll
            for (int t = 0; t < 4; ++t) {
                const int n = t * 16 + m;
                const float p    = e[t] / s;
                const float hard = (n == bi) ? 1.0f : 0.0f;
                Out[orow * Kc + n] = (hard - p) + p;  // straight-through forward
            }
        }
    }
}

extern "C" void kernel_launch(void* const* d_in, const int* in_sizes, int n_in,
                              void* d_out, int out_size, void* d_ws, size_t ws_size,
                              hipStream_t stream) {
    const float* X = (const float*)d_in[0];   // pixel_features [8,65536,256]
    const float* C = (const float*)d_in[1];   // cluster_centers [1,64,256]
    const float* G = (const float*)d_in[2];   // gumbel_noise [8,65536,64]
    float* Out = (float*)d_out;               // [8,65536,64]

    dim3 grid(1024), block(256);
    hipLaunchKernelGGL(kmeans_hard_assign, grid, block, 0, stream, X, C, G, Out);
}
